// MixerDiffAttention_64888365908433
// MI455X (gfx1250) — matrix operand order repaired and checked by reading, not compiled, not run on hardware
//
#include <hip/hip_runtime.h>
#include <hip/hip_bf16.h>
#include <stdint.h>

// ---------------------------------------------------------------------------
// MixerDiffAttention for MI455X (gfx1250, wave32, WMMA + TDM)
//   B=2, T=2048, D=2048, NH=16, HD=128
// Pipeline:
//   1) qkv_gemm      : X @ {Wq,Wk,Wv}  (double-buffered TDM tile loads ->
//                      LDS fp32, bf16 WMMA, f32 accum) -> fp32 ws
//   2) norm_rope_cvt : per-head RMSNorm + RoPE (+ scaler*log_pos*1/sqrt(HD)
//                      folded into Q) -> bf16 ws
//   3) lambda_kernel : lambda_full = exp(sum q1k1) - exp(sum q2k2) + 0.2
//   4) diff_attn     : flash attention (WMMA QK^T and P·V), two passes
//                      (q2/k2 then q1/k1), out = y1 - lambda*y2
// ---------------------------------------------------------------------------

typedef __attribute__((ext_vector_type(16))) __bf16 v16bf;
typedef __attribute__((ext_vector_type(8)))  float  v8f;
typedef __attribute__((ext_vector_type(4)))  unsigned int v4u;
typedef __attribute__((ext_vector_type(8)))  int    v8i;
typedef __attribute__((ext_vector_type(4)))  int    v4i;

constexpr int BB = 2;
constexpr int TT = 2048;
constexpr int DD = 2048;
constexpr int NHh = 16;
constexpr int HDd = 128;

// --- WMMA fragment index helpers (CDNA5 ISA 16-bit layouts, wave32) ---------
// A matrix 16x32 (MxK): lane<16 holds K {0..7,16..23}, lane>=16 holds K {8..15,24..31}
__device__ __forceinline__ int a_kidx(int e, int lane) {
  int v = e >> 1;
  return 2 * v + ((v >= 4) ? 8 : 0) + ((lane >> 4) << 3) + (e & 1);
}
// B matrix 32x16 (KxN): lane<16 holds K {0..7,16..23}, lane>=16 holds K {8..15,24..31}
__device__ __forceinline__ int b_kidx(int e, int lane) {
  return ((lane >> 4) << 3) + (e & 7) + ((e >> 3) << 4);
}

// --- Tensor Data Mover: 2D fp32 tile load Global -> LDS ---------------------
// D# per CDNA5 ISA:
//   group0: count=1, lds_addr, global_addr(57b), type=2
//   group1: data_size=4B, tensor_dim0/1, tile_dim0/1, tensor_dim0_stride
#if defined(__clang_major__) && __clang_major__ >= 23
#define TDM_LOAD(g0, g1, g2, g3) \
  __builtin_amdgcn_tensor_load_to_lds((g0), (g1), (g2), (g3), \
                                      (v8i){0, 0, 0, 0, 0, 0, 0, 0}, 0)
#else
#define TDM_LOAD(g0, g1, g2, g3) \
  __builtin_amdgcn_tensor_load_to_lds((g0), (g1), (g2), (g3), 0)
#endif

__device__ __forceinline__ void tdm_load_2d_f32(
    unsigned lds_addr, const void* gptr,
    unsigned tensor_d0, unsigned tensor_d1, unsigned stride0,
    unsigned tile_d0, unsigned tile_d1)
{
  const unsigned long long ga = (unsigned long long)(uintptr_t)gptr;
  v4u g0;
  g0[0] = 1u;                                   // count = 1 (valid descriptor)
  g0[1] = lds_addr;                             // LDS byte address
  g0[2] = (unsigned)(ga & 0xFFFFFFFFu);         // global_addr[31:0]
  g0[3] = (unsigned)((ga >> 32) & 0x01FFFFFFu)  // global_addr[56:32]
          | (2u << 30);                         // type = 2 ("image")
  v8i g1;
  g1[0] = (int)(2u << 16);                      // data_size = 2 (4 bytes)
  g1[1] = (int)((tensor_d0 & 0xFFFFu) << 16);   // tensor_dim0[15:0] @ bit48
  g1[2] = (int)(((tensor_d0 >> 16) & 0xFFFFu)   // tensor_dim0[31:16]
          | ((tensor_d1 & 0xFFFFu) << 16));     // tensor_dim1[15:0]
  g1[3] = (int)(((tensor_d1 >> 16) & 0xFFFFu)   // tensor_dim1[31:16]
          | ((tile_d0 & 0xFFFFu) << 16));       // tile_dim0
  g1[4] = (int)(tile_d1 & 0xFFFFu);             // tile_dim1 (tile_dim2 = 0)
  g1[5] = (int)stride0;                         // tensor_dim0_stride[31:0]
  g1[6] = 0;                                    // stride0 hi / stride1 lo
  g1[7] = 0;
  const v4i z4 = {0, 0, 0, 0};
  TDM_LOAD(g0, g1, z4, z4);
}

// ---------------------------------------------------------------------------
// Kernel 1: QKV projection GEMM.  M=B*T=4096, K=D=2048, N=D=2048, z selects W.
// Workgroup tile 128x128, 8 waves of 32x64, K-chunk 32.
// Double-buffered TDM: wave 0 issues tile i+1's DMA before draining tile i's
// (TENSORcnt completes in order), so DMA overlaps WMMA compute.
// ---------------------------------------------------------------------------
__global__ __launch_bounds__(256) void qkv_gemm(
    const float* __restrict__ x,
    const float* __restrict__ Wq,
    const float* __restrict__ Wk,
    const float* __restrict__ Wv,
    float* __restrict__ dst_base)
{
  const int z = blockIdx.z;
  const float* __restrict__ W = (z == 0) ? Wq : (z == 1) ? Wk : Wv;
  float* __restrict__ dst = dst_base + (size_t)z * (size_t)(BB * TT) * DD;

  const int n0 = blockIdx.x * 128;
  const int m0 = blockIdx.y * 128;
  const int tid = threadIdx.x;
  const int lane = tid & 31, wid = tid >> 5;
  const int mbase = (wid & 3) * 32;   // 4 waves along M
  const int nbase = (wid >> 2) * 64;  // 2 waves along N

  __shared__ float XsF[2][128][32];   // ping-pong X tiles, 2 x 16 KB
  __shared__ float WtF[2][32][128];   // ping-pong W tiles, 2 x 16 KB

  // LDS byte offsets (flat address low 32 bits == LDS address)
  const unsigned lds_x0 = (unsigned)(uintptr_t)&XsF[0][0][0];
  const unsigned lds_x1 = (unsigned)(uintptr_t)&XsF[1][0][0];
  const unsigned lds_w0 = (unsigned)(uintptr_t)&WtF[0][0][0];
  const unsigned lds_w1 = (unsigned)(uintptr_t)&WtF[1][0][0];

  v8f acc[2][4] = {};

  constexpr int NITER = DD / 32;      // 64 K-chunks

  // Prologue: DMA tile 0 into buffer 0
  if (wid == 0) {
    tdm_load_2d_f32(lds_x0, &x[(size_t)m0 * DD],
                    DD, (unsigned)(BB * TT), DD, 32, 128);
    tdm_load_2d_f32(lds_w0, &W[(size_t)0 * DD + n0],
                    DD, DD, DD, 128, 32);
  }

  for (int i = 0; i < NITER; ++i) {
    const int cur = i & 1;
    if (wid == 0) {
      if (i + 1 < NITER) {
        // Issue next tile into the other buffer (last read in iter i-1,
        // published safe by the end-of-iteration barrier).
        const int k1 = (i + 1) * 32;
        const unsigned lx = (cur == 0) ? lds_x1 : lds_x0;
        const unsigned lw = (cur == 0) ? lds_w1 : lds_w0;
        tdm_load_2d_f32(lx, &x[(size_t)m0 * DD + k1],
                        DD, (unsigned)(BB * TT), DD, 32, 128);
        tdm_load_2d_f32(lw, &W[(size_t)k1 * DD + n0],
                        DD, DD, DD, 128, 32);
        __builtin_amdgcn_s_wait_tensorcnt(2);   // drain oldest pair (tile i)
      } else {
        __builtin_amdgcn_s_wait_tensorcnt(0);   // last tile: drain everything
      }
    }
    __syncthreads();                  // publish buffer `cur` to all waves

    v16bf afrag[2];
    #pragma unroll
    for (int sm = 0; sm < 2; ++sm) {
      const int m = mbase + sm * 16 + (lane & 15);
      #pragma unroll
      for (int j = 0; j < 8; ++j) {   // K-pairs are contiguous -> b64 loads
        const int k = a_kidx(2 * j, lane);
        const float2 f = *reinterpret_cast<const float2*>(&XsF[cur][m][k]);
        afrag[sm][2 * j + 0] = (__bf16)f.x;
        afrag[sm][2 * j + 1] = (__bf16)f.y;
      }
    }
    #pragma unroll
    for (int sn = 0; sn < 4; ++sn) {
      v16bf bfrag;
      const int n = nbase + sn * 16 + (lane & 15);
      #pragma unroll
      for (int e = 0; e < 16; ++e)
        bfrag[e] = (__bf16)WtF[cur][b_kidx(e, lane)][n];
      #pragma unroll
      for (int sm = 0; sm < 2; ++sm)
        acc[sm][sn] = __builtin_amdgcn_wmma_f32_16x16x32_bf16(
            false, afrag[sm], false, bfrag, (short)0, acc[sm][sn], false, false);
    }
    __syncthreads();                  // all waves done reading buffer `cur`
  }

  // Epilogue: C/D layout -> global fp32
  #pragma unroll
  for (int sm = 0; sm < 2; ++sm)
    #pragma unroll
    for (int sn = 0; sn < 4; ++sn)
      #pragma unroll
      for (int r = 0; r < 8; ++r) {
        const int row = m0 + mbase + sm * 16 + ((lane >> 4) << 3) + r;
        const int col = n0 + nbase + sn * 16 + (lane & 15);
        dst[(size_t)row * DD + col] = acc[sm][sn][r];
      }
}

// ---------------------------------------------------------------------------
// Kernel 2: per-head RMSNorm + RoPE; folds softmax_scaler*log(t+1)*1/sqrt(HD)
// into Q; converts Q,K,V to bf16. One block per token, one wave per 2 heads.
// ---------------------------------------------------------------------------
__global__ __launch_bounds__(256) void norm_rope_cvt(
    const float* __restrict__ Qf, const float* __restrict__ Kf,
    const float* __restrict__ Vf, const float* __restrict__ scaler,
    __bf16* __restrict__ qb, __bf16* __restrict__ kbp, __bf16* __restrict__ vbp)
{
  const int token = blockIdx.x;           // b*T + t
  const int t = token & (TT - 1);
  const int lane = threadIdx.x & 31, wid = threadIdx.x >> 5;
  const float logpos = __logf((float)(t + 1));
  const float EPSv = 1.1920929e-7f;
  const float ATTN_SCALE = 0.08838834764831845f;  // 1/sqrt(128)

  // RoPE angles for j = lane and j = lane+32 (j in 0..63)
  float s0, c0, s1, c1;
  {
    const float LN1E4 = 9.210340371976184f;
    const float inv0 = __expf(-((float)(2 * lane) * (1.0f / 128.f)) * LN1E4);
    const float inv1 = __expf(-((float)(2 * (lane + 32)) * (1.0f / 128.f)) * LN1E4);
    __sincosf((float)t * inv0, &s0, &c0);
    __sincosf((float)t * inv1, &s1, &c1);
  }

  #pragma unroll
  for (int hh = 0; hh < 2; ++hh) {
    const int head = wid * 2 + hh;
    const size_t base = (size_t)token * DD + head * HDd;
    #pragma unroll
    for (int which = 0; which < 2; ++which) {   // 0 = Q, 1 = K
      const float* __restrict__ src = which ? (Kf + base) : (Qf + base);
      __bf16* __restrict__ dstp = which ? kbp : qb;
      float a0 = src[lane],      a1 = src[lane + 32];
      float b0 = src[lane + 64], b1 = src[lane + 96];
      float ss = a0 * a0 + a1 * a1 + b0 * b0 + b1 * b1;
      #pragma unroll
      for (int off = 1; off < 32; off <<= 1) ss += __shfl_xor(ss, off, 32);
      const float rr = rsqrtf(ss * (1.0f / HDd) + EPSv);
      a0 *= rr; a1 *= rr; b0 *= rr; b1 *= rr;
      const float sc = which ? 1.0f : scaler[head] * logpos * ATTN_SCALE;
      dstp[base + lane]      = (__bf16)(( a0 * c0 + b0 * s0) * sc);
      dstp[base + lane + 64] = (__bf16)((-a0 * s0 + b0 * c0) * sc);
      dstp[base + lane + 32] = (__bf16)(( a1 * c1 + b1 * s1) * sc);
      dstp[base + lane + 96] = (__bf16)((-a1 * s1 + b1 * c1) * sc);
    }
  }

  // V: straight fp32 -> bf16
  for (int i = threadIdx.x; i < DD; i += 256)
    vbp[(size_t)token * DD + i] = (__bf16)Vf[(size_t)token * DD + i];
}

// ---------------------------------------------------------------------------
// Kernel 3: lambda_full = exp(sum lq1*lk1) - exp(sum lq2*lk2) + 0.2
// ---------------------------------------------------------------------------
__global__ __launch_bounds__(64) void lambda_kernel(
    const float* __restrict__ lq1, const float* __restrict__ lk1,
    const float* __restrict__ lq2, const float* __restrict__ lk2,
    float* __restrict__ lam_out)
{
  const int tid = threadIdx.x;  // 64 = HD/2
  __shared__ float s1[64], s2[64];
  s1[tid] = lq1[tid] * lk1[tid];
  s2[tid] = lq2[tid] * lk2[tid];
  __syncthreads();
  if (tid == 0) {
    float a = 0.f, b = 0.f;
    for (int i = 0; i < 64; ++i) { a += s1[i]; b += s2[i]; }
    lam_out[0] = __expf(a) - __expf(b) + 0.2f;  // LAMBDA_INIT = 0.8 - 0.6*e^0
  }
}

// ---------------------------------------------------------------------------
// Kernel 4: differential flash attention.
// grid = (T/64, NH/2, B), 128 threads = 4 waves; each wave owns 16 queries.
// pass 0: heads 8..15 (q2/k2) -> lambda*y2 into LDS (bf16)
// pass 1: heads 0..7  (q1/k1) -> out = y1 - lambda*y2
// ---------------------------------------------------------------------------
__global__ __launch_bounds__(128) void diff_attn(
    const __bf16* __restrict__ qb, const __bf16* __restrict__ kbp,
    const __bf16* __restrict__ vbp, const float* __restrict__ lamp,
    float* __restrict__ out)
{
  const int lane = threadIdx.x & 31, wave = threadIdx.x >> 5;
  const int qbase = blockIdx.x * 64 + wave * 16;
  const int h = blockIdx.y;
  const int b = blockIdx.z;
  const float lam = lamp[0];

  __shared__ __bf16 Pbuf[4][16 * 32];    // per-wave P scratch
  __shared__ __bf16 Y2[4][16 * 256];     // per-wave lambda*y2

  const int n16 = lane & 15;
  const int hi = lane >> 4;              // 0 or 1 (C-layout half)
  const int kb_end = ((qbase + 15) >> 5) + 1;   // 32-key blocks (causal)

  for (int pass = 0; pass < 2; ++pass) {
    const int head = h + (pass == 0 ? 8 : 0);

    // Q fragments for this wave's 16 rows (K=128 as 4 chunks of 32)
    v16bf qa[4];
    {
      const int qrow = qbase + n16;
      const size_t qoff = ((size_t)(b * TT + qrow) * NHh + head) * HDd;
      #pragma unroll
      for (int c = 0; c < 4; ++c)
        #pragma unroll
        for (int e = 0; e < 16; ++e)
          qa[c][e] = qb[qoff + c * 32 + a_kidx(e, lane)];
    }

    float mrow[8], lrow[8];
    #pragma unroll
    for (int r = 0; r < 8; ++r) { mrow[r] = -3.0e38f; lrow[r] = 0.f; }
    v8f acc[16] = {};

    for (int kblk = 0; kblk < kb_end; ++kblk) {
      const int kbase = kblk * 32;

      // ---- S = Q K^T (two 16x16 tiles over 32 keys) ----
      v8f s2[2] = {};
      #pragma unroll
      for (int nt = 0; nt < 2; ++nt) {
        const int key = kbase + nt * 16 + n16;
        const size_t koff = ((size_t)(b * TT + key) * NHh + head) * HDd;
        #pragma unroll
        for (int c = 0; c < 4; ++c) {
          v16bf kf;
          #pragma unroll
          for (int e = 0; e < 16; ++e)
            kf[e] = kbp[koff + c * 32 + b_kidx(e, lane)];
          s2[nt] = __builtin_amdgcn_wmma_f32_16x16x32_bf16(
              false, qa[c], false, kf, (short)0, s2[nt], false, false);
        }
      }

      // ---- causal mask + online softmax (row stats per r, per half) ----
      #pragma unroll
      for (int r = 0; r < 8; ++r) {
        const int row = qbase + (hi << 3) + r;
        const int j0 = kbase + n16, j1 = j0 + 16;
        float v0 = (j0 <= row) ? s2[0][r] : -1e30f;
        float v1 = (j1 <= row) ? s2[1][r] : -1e30f;
        float mx = fmaxf(v0, v1);
        #pragma unroll
        for (int off = 1; off < 16; off <<= 1)
          mx = fmaxf(mx, __shfl_xor(mx, off, 32));
        const float mnew = fmaxf(mrow[r], mx);
        const float p0 = __expf(v0 - mnew), p1 = __expf(v1 - mnew);
        float sum = p0 + p1;
        #pragma unroll
        for (int off = 1; off < 16; off <<= 1) sum += __shfl_xor(sum, off, 32);
        const float corr = __expf(mrow[r] - mnew);
        lrow[r] = lrow[r] * corr + sum;
        mrow[r] = mnew;
        s2[0][r] = p0; s2[1][r] = p1;
        #pragma unroll
        for (int o = 0; o < 16; ++o) acc[o][r] *= corr;
      }

      // ---- P through LDS: C-layout -> A-fragment layout (bf16) ----
      #pragma unroll
      for (int nt = 0; nt < 2; ++nt)
        #pragma unroll
        for (int r = 0; r < 8; ++r)
          Pbuf[wave][((hi << 3) + r) * 32 + nt * 16 + n16] = (__bf16)s2[nt][r];
      asm volatile("s_wait_dscnt 0" ::: "memory");
      v16bf pa;
      #pragma unroll
      for (int e = 0; e < 16; ++e)
        pa[e] = Pbuf[wave][n16 * 32 + a_kidx(e, lane)];

      // ---- acc += P · V  (V-dim 256 = 16 N-tiles) ----
      #pragma unroll
      for (int o = 0; o < 16; ++o) {
        v16bf vf;
        #pragma unroll
        for (int e = 0; e < 16; ++e) {
          const int key = kbase + b_kidx(e, lane);
          vf[e] = vbp[(size_t)(b * TT + key) * DD + h * 256 + o * 16 + n16];
        }
        acc[o] = __builtin_amdgcn_wmma_f32_16x16x32_bf16(
            false, pa, false, vf, (short)0, acc[o], false, false);
      }
    }

    // ---- finalize pass ----
    if (pass == 0) {
      #pragma unroll
      for (int r = 0; r < 8; ++r) {
        const float s = lam / lrow[r];
        #pragma unroll
        for (int o = 0; o < 16; ++o)
          Y2[wave][((hi << 3) + r) * 256 + o * 16 + n16] =
              (__bf16)(acc[o][r] * s);
      }
      asm volatile("s_wait_dscnt 0" ::: "memory");
    } else {
      #pragma unroll
      for (int r = 0; r < 8; ++r) {
        const float inv = 1.0f / lrow[r];
        const int row = (hi << 3) + r;
        #pragma unroll
        for (int o = 0; o < 16; ++o) {
          const float y2v = (float)Y2[wave][row * 256 + o * 16 + n16];
          out[(size_t)(b * TT + qbase + row) * DD + h * 256 + o * 16 + n16] =
              acc[o][r] * inv - y2v;
        }
      }
    }
  }
}

// ---------------------------------------------------------------------------
extern "C" void kernel_launch(void* const* d_in, const int* in_sizes, int n_in,
                              void* d_out, int out_size, void* d_ws, size_t ws_size,
                              hipStream_t stream) {
  (void)in_sizes; (void)n_in; (void)out_size; (void)ws_size;

  const float* x      = (const float*)d_in[0];
  const float* Wq     = (const float*)d_in[1];
  const float* Wk     = (const float*)d_in[2];
  const float* Wv     = (const float*)d_in[3];
  const float* lq1    = (const float*)d_in[4];
  const float* lk1    = (const float*)d_in[5];
  const float* lq2    = (const float*)d_in[6];
  const float* lk2    = (const float*)d_in[7];
  const float* scaler = (const float*)d_in[8];
  float* out = (float*)d_out;

  const size_t BTD = (size_t)BB * TT * DD;  // 8,388,608
  float*  Qf  = (float*)d_ws;
  float*  Kf  = Qf + BTD;
  float*  Vf  = Kf + BTD;
  __bf16* qb  = (__bf16*)(Vf + BTD);
  __bf16* kbb = qb + BTD;
  __bf16* vbb = kbb + BTD;
  float*  lam = (float*)(vbb + BTD);

  qkv_gemm<<<dim3(DD / 128, (BB * TT) / 128, 3), 256, 0, stream>>>(
      x, Wq, Wk, Wv, Qf);
  norm_rope_cvt<<<dim3(BB * TT), 256, 0, stream>>>(
      Qf, Kf, Vf, scaler, qb, kbb, vbb);
  lambda_kernel<<<dim3(1), 64, 0, stream>>>(lq1, lk1, lq2, lk2, lam);
  diff_attn<<<dim3(TT / 64, NHh / 2, BB), 128, 0, stream>>>(
      qb, kbb, vbb, lam, out);
}
